// TorchFeedForwardNetwork_34849364640195
// MI455X (gfx1250) — matrix operand order repaired and checked by reading, not compiled, not run on hardware
//
#include <hip/hip_runtime.h>
#include <hip/hip_bf16.h>

// ---------------------------------------------------------------------------
// Fused NEAT-MLP forward for MI455X (gfx1250, wave32, WMMA).
//   y = relu(relu(relu(x @ (W1*M1)^T + b1) @ (W2*M2)^T + b2) @ (W3*M3)^T + b3)
// Prep kernel folds mask into weights, converts to bf16 (workspace).
// Main kernel: 64 batch rows per WG (512 threads, 16 waves), activations
// ping-pong in padded LDS as bf16, all matmuls via v_wmma_f32_16x16x32_bf16
// with fp32 accumulation. Padding (+8 halves/row) avoids LDS bank conflicts
// on the 16-row A-fragment loads.
// ---------------------------------------------------------------------------

typedef __attribute__((ext_vector_type(16))) __bf16 v16bf;
typedef __attribute__((ext_vector_type(8)))  __bf16 v8bf;
typedef __attribute__((ext_vector_type(8)))  float  v8f;

#define BT     64   // batch rows per workgroup
#define NTHR   512  // threads per workgroup (16 waves)
#define LDPAD  8    // halves of padding per LDS row (16 B -> 4-bank row skew)

static __device__ __forceinline__ unsigned short f32_to_bf16_rne(float f) {
    unsigned int u = __float_as_uint(f);
    unsigned int r = u + 0x7FFFu + ((u >> 16) & 1u);   // round-to-nearest-even
    return (unsigned short)(r >> 16);
}

static __device__ __forceinline__ unsigned int pack_bf16x2(float lo, float hi) {
    return (unsigned int)f32_to_bf16_rne(lo) |
           ((unsigned int)f32_to_bf16_rne(hi) << 16);
}

// --- prep: Wb[i] = bf16(W[i] * M[i]) ---------------------------------------
__global__ void neat_mask_cvt_kernel(const float* __restrict__ W,
                                     const float* __restrict__ M,
                                     unsigned short* __restrict__ out, int n) {
    int i = blockIdx.x * blockDim.x + threadIdx.x;
    if (i < n) out[i] = f32_to_bf16_rne(W[i] * M[i]);
}

// --- one masked-linear + ReLU layer on a BT-row tile -----------------------
// lds_in : [BT][K+LDPAD] bf16 activations
// Wb     : [N][K] bf16 masked weights (global, row-major over K)
// Output: bf16 into lds_out [BT][N+LDPAD], or fp32 into gout (last layer).
template <int K, int N, bool WRITE_GLOBAL>
static __device__ __forceinline__ void layer_wmma(
    const unsigned short* __restrict__ lds_in,
    unsigned short* __restrict__ lds_out,
    const unsigned short* __restrict__ Wb,
    const float* __restrict__ bias,
    float* __restrict__ gout)
{
    constexpr int LDI = K + LDPAD;
    constexpr int LDO = N + LDPAD;
    const int lane = threadIdx.x & 31;
    const int wave = threadIdx.x >> 5;       // 16 waves
    const int r    = wave & 3;               // 16-row sub-tile within BT=64
    const int q    = wave >> 2;              // quarter of N (0..3)
    constexpr int NT = N / 64;               // 16-col output tiles per wave
    const int nbase = q * (N / 4);

    // A fragment (16x32 bf16): lane holds row M=lane%16;
    // lanes 0-15 -> K chunks {0..7,16..23}, lanes 16-31 -> {8..15,24..31}
    const int rowA = r * 16 + (lane & 15);
    const int kcA  = (lane >> 4) * 8;
    // B fragment (32x16 bf16): lane holds column n=lane%16;
    // lanes 0-15 -> K chunks {0..7,16..23}, lanes 16-31 -> {8..15,24..31}
    const int nB   = lane & 15;
    const int khB  = (lane >> 4) * 8;

    v8f acc[NT] = {};

    for (int k0 = 0; k0 < K; k0 += 32) {
        const unsigned short* ap = lds_in + rowA * LDI + k0 + kcA;
        v8bf alo = *(const v8bf*)ap;          // K = k0+kc .. +8
        v8bf ahi = *(const v8bf*)(ap + 16);   // K = k0+16+kc .. +8
        v16bf a = __builtin_shufflevector(alo, ahi,
                    0,1,2,3,4,5,6,7,8,9,10,11,12,13,14,15);
#pragma unroll
        for (int t = 0; t < NT; t++) {
            const unsigned short* bp =
                Wb + (nbase + t * 16 + nB) * K + k0 + khB;
            v8bf blo = *(const v8bf*)bp;
            v8bf bhi = *(const v8bf*)(bp + 16);
            v16bf b = __builtin_shufflevector(blo, bhi,
                        0,1,2,3,4,5,6,7,8,9,10,11,12,13,14,15);
            acc[t] = __builtin_amdgcn_wmma_f32_16x16x32_bf16(
                false, a, false, b, (short)0, acc[t], false, false);
        }
    }

    // Epilogue: C/D layout: lane = n%16 + 16*(m/8), vgpr v -> m = 8*(lane/16)+v
#pragma unroll
    for (int t = 0; t < NT; t++) {
        const int n  = nbase + t * 16 + (lane & 15);
        const float bv = bias[n];
        const int mb = (lane >> 4) * 8;
#pragma unroll
        for (int v = 0; v < 8; v++) {
            float val = acc[t][v] + bv;
            val = val > 0.0f ? val : 0.0f;            // ReLU (all layers)
            const int row = r * 16 + mb + v;
            if constexpr (WRITE_GLOBAL)
                gout[row * N + n] = val;
            else
                lds_out[row * LDO + n] = f32_to_bf16_rne(val);
        }
    }
}

__global__ __launch_bounds__(NTHR, 1)
void neat_mlp_fused_kernel(const float* __restrict__ x,
                           const float* __restrict__ b1,
                           const float* __restrict__ b2,
                           const float* __restrict__ b3,
                           const unsigned short* __restrict__ Wb1,
                           const unsigned short* __restrict__ Wb2,
                           const unsigned short* __restrict__ Wb3,
                           float* __restrict__ out)
{
    extern __shared__ __align__(16) unsigned short dynsm[];
    unsigned short* sm0 = dynsm;                    // [BT][512+LDPAD] max
    unsigned short* sm1 = dynsm + BT * (512 + LDPAD);

    const long long rb = (long long)blockIdx.x * BT;

    // Stage input tile [BT x 256] f32 -> bf16 LDS (float4 loads, dword stores)
    const float4* xt4 = (const float4*)(x + rb * 256);
    for (int i = threadIdx.x; i < BT * 256 / 4; i += NTHR) {
        float4 f = xt4[i];
        const int e   = i * 4;
        const int row = e >> 8;
        const int col = e & 255;
        unsigned int* dst =
            (unsigned int*)&sm0[row * (256 + LDPAD) + col];
        dst[0] = pack_bf16x2(f.x, f.y);
        dst[1] = pack_bf16x2(f.z, f.w);
    }
    __syncthreads();

    layer_wmma<256, 512, false>(sm0, sm1, Wb1, b1, nullptr);
    __syncthreads();
    layer_wmma<512, 512, false>(sm1, sm0, Wb2, b2, nullptr);
    __syncthreads();
    layer_wmma<512, 256, true >(sm0, nullptr, Wb3, b3, out + rb * 256);
}

// ---------------------------------------------------------------------------
extern "C" void kernel_launch(void* const* d_in, const int* in_sizes, int n_in,
                              void* d_out, int out_size, void* d_ws, size_t ws_size,
                              hipStream_t stream) {
    const float* x  = (const float*)d_in[0];
    const float* W1 = (const float*)d_in[1];
    const float* b1 = (const float*)d_in[2];
    const float* M1 = (const float*)d_in[3];
    const float* W2 = (const float*)d_in[4];
    const float* b2 = (const float*)d_in[5];
    const float* M2 = (const float*)d_in[6];
    const float* W3 = (const float*)d_in[7];
    const float* b3 = (const float*)d_in[8];
    const float* M3 = (const float*)d_in[9];
    // d_in[10..13] are identity index arrays; the graph is a plain MLP.

    unsigned short* Wb1 = (unsigned short*)d_ws;      // 512*256 bf16
    unsigned short* Wb2 = Wb1 + 512 * 256;            // 512*512 bf16
    unsigned short* Wb3 = Wb2 + 512 * 512;            // 256*512 bf16

    neat_mask_cvt_kernel<<<(512 * 256 + 255) / 256, 256, 0, stream>>>(W1, M1, Wb1, 512 * 256);
    neat_mask_cvt_kernel<<<(512 * 512 + 255) / 256, 256, 0, stream>>>(W2, M2, Wb2, 512 * 512);
    neat_mask_cvt_kernel<<<(256 * 512 + 255) / 256, 256, 0, stream>>>(W3, M3, Wb3, 256 * 512);

    const int smem_bytes = 2 * BT * (512 + LDPAD) * (int)sizeof(unsigned short);
    (void)hipFuncSetAttribute((const void*)neat_mlp_fused_kernel,
                              hipFuncAttributeMaxDynamicSharedMemorySize,
                              smem_bytes);

    neat_mlp_fused_kernel<<<65536 / BT, NTHR, smem_bytes, stream>>>(
        x, b1, b2, b3, Wb1, Wb2, Wb3, (float*)d_out);
}
